// QMFNN_81604378624424
// MI455X (gfx1250) — matrix operand order repaired and checked by reading, not compiled, not run on hardware
//
#include <hip/hip_runtime.h>

typedef __attribute__((ext_vector_type(16))) _Float16 v16h;
typedef __attribute__((ext_vector_type(8)))  float    v8f;

#define QBITS 6
#define DIM 64

// ---------------------------------------------------------------------------
// Prep kernel: fold the 12 circuit weights into f16 WMMA B-fragments.
//   frags 0..15 : M_re / M_im  (tt = f>>1: 0..3 re col-tiles, 4..7 im; s = f&1)
//       M_re[c,k] = R[c, Ginv(k)] * cos(theta_k),  M_im with sin(theta_k)
//   frags 16..17: SIGNS^T padded to 64x16 (+-1 exact in f16), for the
//       probability -> PauliZ expectation contraction  out = p @ SIGNS^T
// B-fragment element (lane, j): n=lane&15 (col), h=lane>>4, vi=j>>1, hw=j&1,
//   k = 32*s + 8*h + 2*vi + hw + 8*(vi>>2).
// Total: 18 frags * 32 lanes * 16 halves = 9216 f16 = 18 KB in d_ws.
// ---------------------------------------------------------------------------
__device__ __forceinline__ int cnot_g(int j, int pc, int pt) {
    return j ^ (((j >> pc) & 1) << pt);
}

__global__ void qmfnn_prep_kernel(const float* __restrict__ w,
                                  _Float16* __restrict__ bfrag) {
    int idx = blockIdx.x * blockDim.x + threadIdx.x;
    if (idx >= 18 * 512) return;
    const int j    = idx & 15;
    const int lane = (idx >> 4) & 31;
    const int f    = idx >> 9;          // fragment 0..17

    const int n  = lane & 15;
    const int h  = lane >> 4;
    const int vi = j >> 1;
    const int hw = j & 1;
    const int s  = f & 1;
    const int k  = 32 * s + 8 * h + 2 * vi + hw + 8 * (vi >> 2);   // B-matrix K index

    float val;
    if (f < 16) {
        const int tt  = f >> 1;
        const int col = 16 * (tt & 3) + n;   // output column c (row of M)
        const int mat = tt >> 2;             // 0 = re, 1 = im

        // Ginv(k): inverse of CNOT ring (1,0)(2,1)(3,2)(4,3)(5,4)(0,5);
        // wire w -> bit position 5-w; each CNOT is an involution.
        int t = k;
        t = cnot_g(t, 4, 5);  // CNOT(c=1,t=0)
        t = cnot_g(t, 3, 4);  // CNOT(2,1)
        t = cnot_g(t, 2, 3);  // CNOT(3,2)
        t = cnot_g(t, 1, 2);  // CNOT(4,3)
        t = cnot_g(t, 0, 1);  // CNOT(5,4)
        t = cnot_g(t, 5, 0);  // CNOT(0,5)
        const int j0 = t;

        // R[col, j0] = prod_i RY(w[6+i])[bit_i(col)][bit_i(j0)], RY=[[c,-s],[s,c]]
        float rv = 1.0f;
        #pragma unroll
        for (int i = 0; i < QBITS; ++i) {
            const float ha = 0.5f * w[6 + i];
            const float cc = cosf(ha);
            const float sn = sinf(ha);
            const int bo = (col >> (5 - i)) & 1;
            const int bi = (j0  >> (5 - i)) & 1;
            rv *= bo == 0 ? (bi == 0 ? cc : -sn)
                          : (bi == 0 ? sn :  cc);
        }

        // theta_k from RZ layer: bit=0 -> -w/2, bit=1 -> +w/2
        float th = 0.0f;
        #pragma unroll
        for (int i = 0; i < QBITS; ++i) {
            th += (((k >> (5 - i)) & 1) ? 0.5f : -0.5f) * w[i];
        }
        val = rv * (mat == 0 ? cosf(th) : sinf(th));
    } else {
        // SIGNS^T: B[k][n] = 1 - 2*bit_{wire n}(k) for n<6, else 0
        val = (n < 6) ? (float)(1 - 2 * ((k >> (5 - n)) & 1)) : 0.0f;
    }

    bfrag[(f * 32 + lane) * 16 + j] = (_Float16)val;
}

// ---------------------------------------------------------------------------
// Main kernel: 8 waves/block, 16 batch rows per wave, 18 WMMAs per wave.
//   1) normalize 16 rows of x (f32), pack f16 A-fragments
//   2) 4 col-tiles x {re,im} x 2 K-steps = 16 WMMAs; p = re^2+im^2 -> LDS tile
//   3) transpose-read p from LDS as A-fragments, 2 WMMAs vs the +-1 sign
//      fragments give out[16 rows][6 wires] directly in D layout.
// LDS tiles are wave-private; DS ops are in-order within a wave -> no barrier.
// ---------------------------------------------------------------------------
__global__ void __launch_bounds__(256)
qmfnn_main_kernel(const float* __restrict__ x,
                  const _Float16* __restrict__ bfrag,
                  float* __restrict__ out) {
    __shared__ float pbuf[8][16][68];   // [wave][row][col], pad 68 vs banks

    const int lane = threadIdx.x & 31;
    const int wave = threadIdx.x >> 5;
    const int n = lane & 15;            // A: row within tile; C/D: column
    const int h = lane >> 4;
    const int rowBase = blockIdx.x * 128 + wave * 16;

    const float* xr = x + (size_t)(rowBase + n) * DIM;

    // --- Phase 1a: load half-row of x in the A-fragment K pattern ---
    // runs[s][p] = x[row][32*s + 16*p + 8*h .. +7]
    float runs[2][2][8];
    float ss = 0.0f;
    #pragma unroll
    for (int s = 0; s < 2; ++s) {
        #pragma unroll
        for (int p = 0; p < 2; ++p) {
            const int off = 32 * s + 16 * p + 8 * h;
            const float4 lo = *(const float4*)(xr + off);
            const float4 hi = *(const float4*)(xr + off + 4);
            runs[s][p][0] = lo.x; runs[s][p][1] = lo.y;
            runs[s][p][2] = lo.z; runs[s][p][3] = lo.w;
            runs[s][p][4] = hi.x; runs[s][p][5] = hi.y;
            runs[s][p][6] = hi.z; runs[s][p][7] = hi.w;
            #pragma unroll
            for (int e = 0; e < 8; ++e) ss += runs[s][p][e] * runs[s][p][e];
        }
    }
    // lanes l and l^16 hold complementary halves of the same row
    ss += __shfl_xor(ss, 16, 32);
    const float inv = rsqrtf(ss);   // x/sum then L2-norm == x/||x||2 (x > 0)

    // Pack A fragments (16x32 f16): element j -> vi=j>>1, hw=j&1,
    // K = 32*s + (vi<4 ? 8h+2vi+hw : 16+8h+2(vi-4)+hw)
    v16h a0, a1;
    #pragma unroll
    for (int j = 0; j < 16; ++j) {
        const int vi = j >> 1;
        const int hw = j & 1;
        const int p  = vi >> 2;
        const int e  = 2 * (vi & 3) + hw;
        a0[j] = (_Float16)(runs[0][p][e] * inv);
        a1[j] = (_Float16)(runs[1][p][e] * inv);
    }

    const v16h* Bf = (const v16h*)bfrag;
    float* pb = &pbuf[wave][0][0];

    // --- Phase 2: Y_re, Y_im tiles; p = re^2 + im^2 -> LDS ---
    #pragma unroll
    for (int t = 0; t < 4; ++t) {
        const v16h br0 = Bf[((t)     * 2 + 0) * 32 + lane];
        const v16h br1 = Bf[((t)     * 2 + 1) * 32 + lane];
        const v16h bi0 = Bf[((4 + t) * 2 + 0) * 32 + lane];
        const v16h bi1 = Bf[((4 + t) * 2 + 1) * 32 + lane];

        v8f cre = {};
        cre = __builtin_amdgcn_wmma_f32_16x16x32_f16(false, a0, false, br0,
                                                     (short)0, cre, false, false);
        cre = __builtin_amdgcn_wmma_f32_16x16x32_f16(false, a1, false, br1,
                                                     (short)0, cre, false, false);
        v8f cim = {};
        cim = __builtin_amdgcn_wmma_f32_16x16x32_f16(false, a0, false, bi0,
                                                     (short)0, cim, false, false);
        cim = __builtin_amdgcn_wmma_f32_16x16x32_f16(false, a1, false, bi1,
                                                     (short)0, cim, false, false);

        // C/D layout: lane holds col 16t+n, VGPR r holds row r+8h
        #pragma unroll
        for (int r = 0; r < 8; ++r) {
            const float p = cre[r] * cre[r] + cim[r] * cim[r];
            pb[(r + 8 * h) * 68 + 16 * t + n] = p;
        }
    }

    // --- Phase 3: transpose-read p as A-fragments (same-wave LDS, in-order) ---
    const float* prow = &pbuf[wave][n][0];   // row n, 272B stride (16B aligned)
    float rr[2][2][8];
    #pragma unroll
    for (int s = 0; s < 2; ++s) {
        #pragma unroll
        for (int p = 0; p < 2; ++p) {
            const int off = 32 * s + 16 * p + 8 * h;
            const float4 lo = *(const float4*)(prow + off);
            const float4 hi = *(const float4*)(prow + off + 4);
            rr[s][p][0] = lo.x; rr[s][p][1] = lo.y;
            rr[s][p][2] = lo.z; rr[s][p][3] = lo.w;
            rr[s][p][4] = hi.x; rr[s][p][5] = hi.y;
            rr[s][p][6] = hi.z; rr[s][p][7] = hi.w;
        }
    }
    v16h pa0, pa1;
    #pragma unroll
    for (int j = 0; j < 16; ++j) {
        const int vi = j >> 1;
        const int hw = j & 1;
        const int p  = vi >> 2;
        const int e  = 2 * (vi & 3) + hw;
        pa0[j] = (_Float16)rr[0][p][e];
        pa1[j] = (_Float16)rr[1][p][e];
    }

    const v16h bs0 = Bf[16 * 32 + lane];
    const v16h bs1 = Bf[17 * 32 + lane];
    v8f d = {};
    d = __builtin_amdgcn_wmma_f32_16x16x32_f16(false, pa0, false, bs0,
                                               (short)0, d, false, false);
    d = __builtin_amdgcn_wmma_f32_16x16x32_f16(false, pa1, false, bs1,
                                               (short)0, d, false, false);

    // D layout: lane n = wire w (n<6 valid), VGPR r = row r+8h
    if (n < 6) {
        #pragma unroll
        for (int r = 0; r < 8; ++r) {
            out[(size_t)(rowBase + r + 8 * h) * 6 + n] = d[r];
        }
    }
}

extern "C" void kernel_launch(void* const* d_in, const int* in_sizes, int n_in,
                              void* d_out, int out_size, void* d_ws, size_t ws_size,
                              hipStream_t stream) {
    const float* x = (const float*)d_in[0];   // [BATCH, 64] f32
    const float* w = (const float*)d_in[1];   // [12] f32
    float* out = (float*)d_out;               // [BATCH, 6] f32
    _Float16* bf = (_Float16*)d_ws;           // 18 KB B-fragment table

    const int batch = in_sizes[0] / DIM;

    // 9216 fragment halves, 256 threads/block
    qmfnn_prep_kernel<<<36, 256, 0, stream>>>(w, bf);

    // 128 rows per block (8 waves x 16 rows); BATCH = 262144 is divisible
    qmfnn_main_kernel<<<batch / 128, 256, 0, stream>>>(x, bf, out);
}
